// QuantLinear_15066745274475
// MI455X (gfx1250) — matrix operand order repaired and checked by reading, not compiled, not run on hardware
//
#include <hip/hip_runtime.h>
#include <hip/hip_bf16.h>
#include <stdint.h>

typedef __attribute__((ext_vector_type(16))) _Float16 v16h;
typedef __attribute__((ext_vector_type(8)))  _Float16 v8h;
typedef __attribute__((ext_vector_type(4)))  _Float16 v4h;
typedef __attribute__((ext_vector_type(2)))  _Float16 h2;
typedef __attribute__((ext_vector_type(2)))  __fp16   pk2;   // cvt_pkrtz result type
typedef __attribute__((ext_vector_type(8)))  float    v8f;

#define K_DIM 4096
#define N_DIM 11008
#define LDSH  40          // halves per LDS row: 32 + 8 pad (80B, 16B-aligned)

// Raw LDS byte offset of a pointer into a __shared__ array (addrspace(3) cast).
__device__ __forceinline__ uint32_t lds_byte_off(const void* p) {
    return (uint32_t)(uintptr_t)(const __attribute__((address_space(3))) void*)p;
}

// Pack 4 floats -> 4 f16 (two v_cvt_pk_f16_f32), returned as v4h.
__device__ __forceinline__ v4h pack4(float a, float b, float c, float d) {
    union { pk2 p[2]; v4h v; } u;
    u.p[0] = __builtin_amdgcn_cvt_pkrtz(a, b);
    u.p[1] = __builtin_amdgcn_cvt_pkrtz(c, d);
    return u.v;
}

__global__ void __launch_bounds__(256)
gptq_wmma_kernel(const float* __restrict__ x,
                 const float* __restrict__ scales,
                 const float* __restrict__ bias,
                 const int*   __restrict__ qweight,
                 const int*   __restrict__ qzeros,
                 float*       __restrict__ out) {
    __shared__ _Float16 sA[2][128 * LDSH];   // X tile as f16, double-buffered
    __shared__ int      sQ[2][4 * 128];      // packed-weight tile, double-buffered

    const int tid      = threadIdx.x;
    const int lane     = tid & 31;
    const int wv       = tid >> 5;       // 0..7 : wave owns 16 N cols, all 128 M rows
    const int laneHalf = lane >> 4;      // 0: lanes 0-15, 1: lanes 16-31
    const int lmod     = lane & 15;

    const int rowBase = blockIdx.y * 128;
    const int colBase = blockIdx.x * 128;
    const int nCol    = colBase + wv * 16 + lmod;   // this lane's output column
    const int cLoc    = wv * 16 + lmod;             // column within block tile

    const int xRow = tid >> 3;           // 0..31 (+32*i), X staging row
    const int xC4  = (tid & 7) * 4;      // X staging col (floats)
    const int qRow = tid >> 6;           // 0..3, qweight staging row
    const int qCol = (tid & 63) * 2;     // qweight staging col (2 int32 per thread)

    v8f acc[8];
#pragma unroll
    for (int sm = 0; sm < 8; ++sm) acc[sm] = (v8f)0.0f;

    // ---------------- Prologue: stage tile 0 ----------------
    {
        const float* src = x + (size_t)rowBase * K_DIM;
#pragma unroll
        for (int i = 0; i < 4; ++i) {
            const int row = xRow + 32 * i;
            float4 v = *(const float4*)(src + (size_t)row * K_DIM + xC4);
            *(v4h*)&sA[0][row * LDSH + xC4] = pack4(v.x, v.y, v.z, v.w);
        }
        const int* qsrc = qweight + (size_t)colBase;   // kRowBase = 0
        uint32_t goff = (uint32_t)((qRow * N_DIM + qCol) * 4);
        uint32_t loff = lds_byte_off(&sQ[0][qRow * 128 + qCol]);
        asm volatile("global_load_async_to_lds_b64 %0, %1, %2 offset:0"
                     :: "v"(loff), "v"(goff), "s"(qsrc) : "memory");
    }
    asm volatile("s_wait_asynccnt 0x0" ::: "memory");
    __syncthreads();

    int buf = 0;
    for (int g = 0; g < 32; ++g) {                 // 32 quant groups of 128
        // Per-group f16 dequant coefficients for this lane's single column:
        //   deq = fma(0x6400|nib, s, -s*(1025+z))   since (0x6400|nib) == 1024+nib
        const float s  = scales[(size_t)g * N_DIM + nCol];
        const uint32_t qz = (uint32_t)qzeros[(size_t)g * (N_DIM / 8) + (nCol >> 3)];
        const float zq = (float)((qz >> (4 * (nCol & 7))) & 15u);
        const _Float16 hs = (_Float16)s;
        const _Float16 hc = (_Float16)(-s * (1025.0f + zq));
        h2 s2, c2;
        s2[0] = hs; s2[1] = hs;
        c2[0] = hc; c2[1] = hc;

#pragma unroll
        for (int kk = 0; kk < 4; ++kk) {           // 4 WMMA k-steps per group
            const int kBase = g * 128 + kk * 32;
            const int it    = g * 4 + kk;
            const bool hasNext = (it + 1 < 128);

            // ---- Start fetching next X tile into registers (hides latency) ----
            float4 xr[4];
            if (hasNext) {
                const float* src = x + (size_t)rowBase * K_DIM + (kBase + 32);
#pragma unroll
                for (int i = 0; i < 4; ++i) {
                    const int row = xRow + 32 * i;
                    xr[i] = *(const float4*)(src + (size_t)row * K_DIM + xC4);
                }
                // ---- Async DMA of next qweight tile into the other buffer ----
                const int* qsrc = qweight + (size_t)((kBase + 32) >> 3) * N_DIM + colBase;
                uint32_t goff = (uint32_t)((qRow * N_DIM + qCol) * 4);
                uint32_t loff = lds_byte_off(&sQ[buf ^ 1][qRow * 128 + qCol]);
                asm volatile("global_load_async_to_lds_b64 %0, %1, %2 offset:0"
                             :: "v"(loff), "v"(goff), "s"(qsrc) : "memory");
                __builtin_prefetch(qsrc + (size_t)4 * N_DIM + cLoc, 0, 1);
            }

            // ---- B fragment: perm-based packed-f16 dequant from LDS ----
            // Lanes 0-15 hold K 0..7,16..23; lanes 16-31 hold K 8..15,24..31.
            // Each block column is dequantized exactly once (no wave redundancy).
            const int rsel = laneHalf;
            union BF { v16h v; h2 p[8]; } bf;
            {
                uint32_t q[2];
                q[0] = (uint32_t)sQ[buf][(rsel    ) * 128 + cLoc];
                q[1] = (uint32_t)sQ[buf][(rsel + 2) * 128 + cLoc];
#pragma unroll
                for (int d = 0; d < 2; ++d) {
                    const uint32_t e = q[d] & 0x0F0F0F0Fu;          // nibbles 0,2,4,6
                    const uint32_t o = (q[d] >> 4) & 0x0F0F0F0Fu;   // nibbles 1,3,5,7
#pragma unroll
                    for (int pp = 0; pp < 4; ++pp) {
                        const uint32_t sel = 0x0C000C00u | ((uint32_t)pp << 16)
                                           | (uint32_t)(4 + pp);
                        union PW { uint32_t u; h2 h; } pw;
                        pw.u = __builtin_amdgcn_perm(e, o, sel) | 0x64006400u;
                        bf.p[d * 4 + pp] = pw.h * s2 + c2;          // v_pk_fma_f16
                    }
                }
            }

            // ---- 8 A fragments (pure b128 LDS loads) + 8 WMMA ----
#pragma unroll
            for (int sm = 0; sm < 8; ++sm) {
                const int rloc = sm * 16 + lmod;
                const _Float16* rowp = &sA[buf][rloc * LDSH + laneHalf * 8];
                v8h a0 = *(const v8h*)(rowp);        // K 0..7  (or 8..15)
                v8h a1 = *(const v8h*)(rowp + 16);   // K 16..23 (or 24..31)
                v16h afrag = __builtin_shufflevector(a0, a1,
                    0,1,2,3,4,5,6,7, 8,9,10,11,12,13,14,15);
                acc[sm] = __builtin_amdgcn_wmma_f32_16x16x32_f16(
                    false, afrag, false, bf.v,
                    (short)0, acc[sm], false, false);
            }

            // ---- Convert + store next X tile into the other buffer ----
            if (hasNext) {
#pragma unroll
                for (int i = 0; i < 4; ++i) {
                    const int row = xRow + 32 * i;
                    *(v4h*)&sA[buf ^ 1][row * LDSH + xC4] =
                        pack4(xr[i].x, xr[i].y, xr[i].z, xr[i].w);
                }
            }

            // Async DMA of next q-tile must land; then sync whole workgroup.
            asm volatile("s_wait_asynccnt 0x0" ::: "memory");
            __syncthreads();
            buf ^= 1;
        }
    }

    // ---- Epilogue: bias add + store (C layout: VGPR r -> M = r / r+8) ----
    {
        const float bv = bias[nCol];
#pragma unroll
        for (int sm = 0; sm < 8; ++sm) {
            const int mBase = rowBase + sm * 16 + laneHalf * 8;
#pragma unroll
            for (int r = 0; r < 8; ++r) {
                out[(size_t)(mBase + r) * N_DIM + nCol] = acc[sm][r] + bv;
            }
        }
    }
}

extern "C" void kernel_launch(void* const* d_in, const int* in_sizes, int n_in,
                              void* d_out, int out_size, void* d_ws, size_t ws_size,
                              hipStream_t stream) {
    const float* x       = (const float*)d_in[0];
    const float* scales  = (const float*)d_in[1];
    const float* bias    = (const float*)d_in[2];
    const int*   qweight = (const int*)d_in[3];
    const int*   qzeros  = (const int*)d_in[4];
    // d_in[5] = g_idx: contiguous groups of 128, recomputed analytically.
    float* out = (float*)d_out;

    dim3 grid(N_DIM / 128, 8192 / 128);  // 86 x 64
    dim3 block(256);
    gptq_wmma_kernel<<<grid, block, 0, stream>>>(x, scales, bias, qweight, qzeros, out);
}
